// GRU_63720134803698
// MI455X (gfx1250) — matrix operand order repaired and checked by reading, not compiled, not run on hardware
//
#include <hip/hip_runtime.h>

// GRU: T=512, B=256, I=256, H=256.
// gate[b,n] = sum_{k<H} h[b,k]*W[n,k] + sum_{k<I} x[b,k]*W[n,H+k] + bias[n]
// Batch rows are independent across time -> 16 persistent workgroups,
// each owns a 16-row batch tile for all T steps. bf16 WMMA, f32 accum.
// v4: launder an integer OFFSET (not the pointer) so weight loads keep their
// global address-space (global_load_b128, LOADcnt-only) instead of falling
// back to flat_load (which couples to DScnt and the LDS pipe), while still
// defeating LICM hoist+spill of the T-invariant weight fragments.

#define T_STEPS 512
#define BATCH   256
#define ISZ     256
#define HSZ     256
#define KW      512      // I+H: weight row length (row-major [H][I+H])
#define LDP     264      // padded LDS row stride in bf16 elems (breaks bank conflicts)

typedef __attribute__((ext_vector_type(16))) __bf16       v16bf;
typedef __attribute__((ext_vector_type(8)))  float        v8f;
typedef __attribute__((ext_vector_type(8)))  unsigned int v8u;
typedef __attribute__((ext_vector_type(4)))  unsigned int v4u;

// assemble a 16-element bf16 fragment from two 16-byte chunks (16-B aligned)
static __device__ __forceinline__ v16bf frag_merge(const __bf16* lo, const __bf16* hi) {
    v4u a = *(const v4u*)lo;
    v4u b = *(const v4u*)hi;
    v8u w = __builtin_shufflevector(a, b, 0, 1, 2, 3, 4, 5, 6, 7);
    return __builtin_bit_cast(v16bf, w);
}

// A-matrix 16x32 bf16 fragment from padded LDS buffer (row-major, stride LDP).
static __device__ __forceinline__ v16bf afrag(const __bf16* buf, int lane, int ks) {
    int m  = lane & 15;
    int k0 = ks * 32 + ((lane >> 4) << 4);
    const __bf16* p = buf + m * LDP + k0;
    return frag_merge(p, p + 8);
}

// B-matrix 32x16 bf16 fragment from global row-major [N][KW] weights.
static __device__ __forceinline__ v16bf bfrag_g(const __bf16* w, int n, int lane, int kbase) {
    int k0 = kbase + ((lane >> 4) << 4);
    const __bf16* p = w + n * KW + k0;
    return frag_merge(p, p + 8);
}

// B-matrix fragment from padded LDS weight copy (stride LDP).
static __device__ __forceinline__ v16bf bfrag_l(const __bf16* w, int n, int lane, int ks) {
    int k0 = ks * 32 + ((lane >> 4) << 4);
    const __bf16* p = w + n * LDP + k0;
    return frag_merge(p, p + 8);
}

static __device__ __forceinline__ v8f wmma_bf16(v16bf a, v16bf b, v8f c) {
    return __builtin_amdgcn_wmma_f32_16x16x32_bf16(false, a, false, b, (short)0, c, false, false);
}

static __device__ __forceinline__ float fast_sigmoid(float x) {
    float e = __builtin_amdgcn_exp2f(-1.442695041f * x);
    return __builtin_amdgcn_rcpf(1.0f + e);
}
static __device__ __forceinline__ float fast_tanh(float x) {
    float e = __builtin_amdgcn_exp2f(2.885390082f * x);
    return 1.0f - 2.0f * __builtin_amdgcn_rcpf(e + 1.0f);
}

// ---------------- phase 0: convert weights f32 -> bf16 (once) ----------------
__global__ void cvt_weights(const float* __restrict__ a, const float* __restrict__ b,
                            const float* __restrict__ c,
                            __bf16* __restrict__ oa, __bf16* __restrict__ ob,
                            __bf16* __restrict__ oc) {
    int i = blockIdx.x * blockDim.x + threadIdx.x;
    if (i < HSZ * KW) {
        oa[i] = (__bf16)a[i];
        ob[i] = (__bf16)b[i];
        oc[i] = (__bf16)c[i];
    }
}

// ---------------- phase 1: persistent GRU over T ----------------
__global__ __launch_bounds__(512)
void gru_persistent(const float* __restrict__ x, const float* __restrict__ h0,
                    const __bf16* __restrict__ wz, const __bf16* __restrict__ wr,
                    const __bf16* __restrict__ wc,
                    const float* __restrict__ bz, const float* __restrict__ br,
                    const float* __restrict__ bc,
                    float* __restrict__ out) {
    __shared__ __bf16 WcL[256 * LDP];   // candidate weight, h-part, 132 KB
    __shared__ __bf16 hB [16 * LDP];    // h_{t-1} tile, bf16 (A-matrix source)
    __shared__ __bf16 rhB[16 * LDP];    // r*h tile, bf16
    __shared__ __bf16 xB [16 * LDP];    // x_t tile, bf16

    const int tid  = threadIdx.x;       // 0..511 -> 16 waves
    const int lane = tid & 31;
    const int wave = tid >> 5;
    const int m0   = blockIdx.x * 16;   // batch tile origin
    const int n0   = wave * 16;         // output-column strip
    const int nl   = lane & 15;
    const int n    = n0 + nl;           // global output column for this lane
    const int mb   = (lane >> 4) << 3;  // C-layout row offset (0 or 8)

    // stage Wc h-part (256x256 bf16) into padded LDS
    for (int i = tid; i < 256 * 256; i += 512) {
        int r = i >> 8, k = i & 255;
        WcL[r * LDP + k] = wc[r * KW + k];
    }

    // init h tile: f32 master copy in registers (exactly the WMMA C layout)
    v8f hT;
#pragma unroll
    for (int v = 0; v < 8; ++v) {
        int m = v + mb;
        float hv = h0[(m0 + m) * HSZ + n];
        hT[v] = hv;
        hB[m * LDP + n] = (__bf16)hv;
    }

    const float bzv = bz[n], brv = br[n], bcv = bc[n];

    __syncthreads();

    for (int t = 0; t < T_STEPS; ++t) {
        // Launder an integer offset (SGPR) + LDS offset (VGPR): the volatile
        // asm makes the addresses opaque per-iteration so LICM cannot hoist
        // the T-invariant weight-fragment loads (which caused scratch spills),
        // while GEPs on the original pointers keep addrspace(1) provenance so
        // the backend emits global_load_b128 (LOADcnt-only), not flat_load.
        // Weights (768 KB bf16 total) are L2-resident on the 192 MB L2.
        unsigned int goff = 0;
        int wcoff = 0;
        asm volatile("" : "+s"(goff), "+v"(wcoff));
        const __bf16* wzp = wz + goff;
        const __bf16* wrp = wr + goff;
        const __bf16* wcp = wc + goff;

        // stage x_t tile [16 x 256] f32 -> bf16 LDS (x read once from HBM total)
        {
            int base = tid * 8;
            int r = base >> 8, c = base & 255;
            const float* xs = x + ((size_t)t * BATCH + m0 + r) * ISZ + c;
#pragma unroll
            for (int j = 0; j < 8; ++j)
                xB[r * LDP + c + j] = (__bf16)xs[j];
        }
        __syncthreads();   // xB ready; hB from previous step visible

        // z and r gates: K=512; A from LDS, B streamed from L2-resident bf16 weights
        v8f accZ, accR;
#pragma unroll
        for (int v = 0; v < 8; ++v) { accZ[v] = bzv; accR[v] = brv; }

#pragma unroll
        for (int ks = 0; ks < 8; ++ks) {              // K = 0..255 (h)
            v16bf a = afrag(hB, lane, ks);
            accZ = wmma_bf16(a, bfrag_g(wzp, n, lane, ks * 32), accZ);
            accR = wmma_bf16(a, bfrag_g(wrp, n, lane, ks * 32), accR);
        }
#pragma unroll
        for (int ks = 0; ks < 8; ++ks) {              // K = 256..511 (x)
            v16bf a = afrag(xB, lane, ks);
            accZ = wmma_bf16(a, bfrag_g(wzp, n, lane, 256 + ks * 32), accZ);
            accR = wmma_bf16(a, bfrag_g(wrp, n, lane, 256 + ks * 32), accR);
        }

        float zv[8], rv[8];
#pragma unroll
        for (int v = 0; v < 8; ++v) {
            zv[v] = fast_sigmoid(accZ[v]);
            rv[v] = fast_sigmoid(accR[v]);
            rhB[(v + mb) * LDP + n] = (__bf16)(rv[v] * hT[v]);
        }
        __syncthreads();   // rhB ready; all waves done reading hB

        // candidate: K=512 (r*h part weights from LDS, x-part streamed)
        v8f accC;
#pragma unroll
        for (int v = 0; v < 8; ++v) accC[v] = bcv;
#pragma unroll
        for (int ks = 0; ks < 8; ++ks)
            accC = wmma_bf16(afrag(rhB, lane, ks), bfrag_l(WcL + wcoff, n, lane, ks), accC);
#pragma unroll
        for (int ks = 0; ks < 8; ++ks)
            accC = wmma_bf16(afrag(xB, lane, ks), bfrag_g(wcp, n, lane, 256 + ks * 32), accC);

        // h_t = (1-z)*h + z*tanh(c); write hAll, refresh LDS h for next step
#pragma unroll
        for (int v = 0; v < 8; ++v) {
            int m = v + mb;
            float hv = (1.0f - zv[v]) * hT[v] + zv[v] * fast_tanh(accC[v]);
            hT[v] = hv;
            out[((size_t)t * BATCH + m0 + m) * HSZ + n] = hv;
            hB[m * LDP + n] = (__bf16)hv;
        }
        if (t == T_STEPS - 1) {
#pragma unroll
            for (int v = 0; v < 8; ++v) {
                int m = v + mb;
                out[(size_t)T_STEPS * BATCH * HSZ + (size_t)(m0 + m) * HSZ + n] = hT[v];
            }
        }
        __syncthreads();   // hB/xB consumers done before next step overwrites
    }
}

extern "C" void kernel_launch(void* const* d_in, const int* in_sizes, int n_in,
                              void* d_out, int out_size, void* d_ws, size_t ws_size,
                              hipStream_t stream) {
    (void)in_sizes; (void)n_in; (void)out_size; (void)ws_size;

    const float* x   = (const float*)d_in[0];
    const float* h0  = (const float*)d_in[1];
    const float* wzF = (const float*)d_in[2];
    const float* bz  = (const float*)d_in[3];
    const float* wrF = (const float*)d_in[4];
    const float* br  = (const float*)d_in[5];
    const float* wcF = (const float*)d_in[6];
    const float* bc  = (const float*)d_in[7];
    float* out = (float*)d_out;

    // workspace: 3 bf16 weight copies, 256 KB each (768 KB total)
    __bf16* wzB = (__bf16*)d_ws;
    __bf16* wrB = wzB + HSZ * KW;
    __bf16* wcB = wrB + HSZ * KW;

    cvt_weights<<<(HSZ * KW + 255) / 256, 256, 0, stream>>>(wzF, wrF, wcF, wzB, wrB, wcB);
    gru_persistent<<<BATCH / 16, 512, 0, stream>>>(x, h0, wzB, wrB, wcB, bz, br, bc, out);
}
